// RegisterTiledHyperbolicAttention_16870631539338
// MI455X (gfx1250) — matrix operand order repaired and checked by reading, not compiled, not run on hardware
//
#include <hip/hip_runtime.h>
#include <hip/hip_bf16.h>

// ---------------------------------------------------------------------------
// Hyperbolic attention, fully WMMA-based for gfx1250 (CDNA5, wave32).
//   B=2, N=2048, D=1024, H=16, Dh=64, curvature c=1.
// Pipeline (all on `stream`):
//   1) cvt:    f32 -> f16 copies of x and the 4 weights (workspace)
//   2) gemm16<0>: Q = x @ wq^T   (f16 out, row-major)
//      gemm16<0>: K = x @ wk^T   (f16 out, row-major)
//      gemm16<1>: V = x @ wv^T   (f16 out, TRANSPOSED vT[col][m] for P·V B-frags)
//   3) norms:  qn[m,h] = ||q_head||^2, kn[m,h] = ||k_head||^2  (f32)
//   4) attn:   flash-style; S = Q·K^T via v_wmma; p = 1/(z+sqrt(z^2-1))
//              (= exp(-arccosh(z)), no log/exp needed); P·V via v_wmma with
//              P staged through wave-private LDS (A-fragment relayout).
//              Uses v_rcp_f32 / v_sqrt_f32 (TRANS pipe) for the N^2 epilogue.
//   5) gemm16<2>: out = O @ wo^T (f32 out -> d_out)
// ---------------------------------------------------------------------------

typedef __attribute__((ext_vector_type(4)))  _Float16 v4h;
typedef __attribute__((ext_vector_type(8)))  _Float16 v8h;
typedef __attribute__((ext_vector_type(16))) _Float16 v16h;
typedef __attribute__((ext_vector_type(8)))  float    v8f;

#define WMMA_F16(a, b, c) \
  __builtin_amdgcn_wmma_f32_16x16x32_f16(false, (a), false, (b), (short)0, (c), false, false)

static __device__ __forceinline__ v16h cat16(v8h lo, v8h hi) {
  return __builtin_shufflevector(lo, hi, 0,1,2,3,4,5,6,7,8,9,10,11,12,13,14,15);
}
static __device__ __forceinline__ v8f zero8f() {
  v8f z;
#pragma unroll
  for (int i = 0; i < 8; ++i) z[i] = 0.0f;
  return z;
}

// Problem constants
#define BSZ   2
#define NSEQ  2048
#define DMOD  1024
#define NH    16
#define DH    64
#define MROW  4096           // B*N

// Workspace layout (offsets in _Float16 elements)
#define HX    ((size_t)0)            // x f16:      4096*1024
#define HWQ   ((size_t)4194304)      // wq f16:     1024*1024
#define HWK   ((size_t)5242880)
#define HWV   ((size_t)6291456)
#define HWO   ((size_t)7340032)
#define HQ    ((size_t)8388608)      // Q f16 [m][1024]
#define HK    ((size_t)12582912)     // K f16 [m][1024]
#define HVT   ((size_t)16777216)     // V^T f16 [col][m] (col=h*64+d, m=b*2048+n)
#define HO    ((size_t)20971520)     // O f16 [m][1024]
#define H_END ((size_t)25165824)
#define QN_BYTE  (H_END * 2)                  // f32 qn[m*16+h], 65536 floats
#define KN_BYTE  (QN_BYTE + 65536 * 4)        // f32 kn[m*16+h]

// ------------------------------- 1) convert --------------------------------
__global__ __launch_bounds__(256)
void cvt_f32_to_f16(const float* __restrict__ x,  const float* __restrict__ wq,
                    const float* __restrict__ wk, const float* __restrict__ wv,
                    const float* __restrict__ wo, _Float16* __restrict__ ws) {
  size_t idx = (size_t)blockIdx.x * blockDim.x + threadIdx.x;  // 2M threads
  size_t i4 = idx * 4;                                          // 8M elements
  const float* src;
  size_t soff, doff;
  if (i4 < HWQ) {                       // x region (4M elements)
    src = x; soff = i4; doff = HX + i4;
  } else {                              // weight region (4 x 1M)
    size_t j = i4 - HWQ;
    int w = (int)(j >> 20);
    size_t r = j & 1048575u;
    src = (w == 0) ? wq : (w == 1) ? wk : (w == 2) ? wv : wo;
    soff = r; doff = HWQ + j;
  }
  float4 f = *(const float4*)(src + soff);
  v4h o;
  o[0] = (_Float16)f.x; o[1] = (_Float16)f.y;
  o[2] = (_Float16)f.z; o[3] = (_Float16)f.w;
  *(v4h*)(ws + doff) = o;
}

// ------------------------------- 2) GEMMs ----------------------------------
// Y[m, c] = sum_k A[m,k] * W[c,k]   (A: [4096][1024] f16, W: [1024][1024] f16)
// MODE 0: f16 out row-major [m][1024]
// MODE 1: f16 out transposed [c][4096]  (packed v8h store: rows are contiguous)
// MODE 2: f32 out row-major [m][1024]  (final output)
template <int MODE>
__global__ __launch_bounds__(256)
void gemm16(const _Float16* __restrict__ A, const _Float16* __restrict__ W,
            void* __restrict__ Out) {
  const int K = DMOD;
  int wave = threadIdx.x >> 5, lane = threadIdx.x & 31;
  int lo = lane & 15, hh = lane >> 4;               // hh: low/high half of wave
  int gw = blockIdx.x * 8 + wave;                   // 4096 wave tiles
  int m0 = (gw >> 4) * 16;                          // 16-row tile
  int c0 = (gw & 15) * 64;                          // 64-col tile (4 WMMA tiles)

  v8f c[4];
#pragma unroll
  for (int t = 0; t < 4; ++t) c[t] = zero8f();

  const _Float16* Abase = A + (size_t)(m0 + lo) * K;

  for (int k0 = 0; k0 < K; k0 += 32) {
    // A fragment (16x32): lane row = lo; elements 0-15 @ K=k0+hh*16 (contiguous)
    int ka = k0 + hh * 16;
    v8h alo = *(const v8h*)(Abase + ka);
    v8h ahi = *(const v8h*)(Abase + ka + 8);
    v16h a = cat16(alo, ahi);
    int kb = k0 + hh * 16;                          // B frag: K 0-15 / 16-31
#pragma unroll
    for (int t = 0; t < 4; ++t) {
      int col = c0 + t * 16 + lo;                   // B col = output col (W row)
      v16h b = *(const v16h*)(W + (size_t)col * K + kb);
      c[t] = WMMA_F16(a, b, c[t]);
    }
  }

  if (MODE == 0) {
    _Float16* O = (_Float16*)Out;
#pragma unroll
    for (int t = 0; t < 4; ++t)
#pragma unroll
      for (int r = 0; r < 8; ++r)
        O[(size_t)(m0 + hh * 8 + r) * DMOD + c0 + t * 16 + lo] = (_Float16)c[t][r];
  } else if (MODE == 1) {
    _Float16* O = (_Float16*)Out;                   // vT[col][MROW]
#pragma unroll
    for (int t = 0; t < 4; ++t) {
      v8h p;
#pragma unroll
      for (int r = 0; r < 8; ++r) p[r] = (_Float16)c[t][r];
      *(v8h*)(O + (size_t)(c0 + t * 16 + lo) * MROW + m0 + hh * 8) = p;
    }
  } else {
    float* O = (float*)Out;
#pragma unroll
    for (int t = 0; t < 4; ++t)
#pragma unroll
      for (int r = 0; r < 8; ++r)
        O[(size_t)(m0 + hh * 8 + r) * DMOD + c0 + t * 16 + lo] = c[t][r];
  }
}

// ------------------------------- 3) norms ----------------------------------
__global__ __launch_bounds__(256)
void head_norms(const _Float16* __restrict__ q, const _Float16* __restrict__ k,
                float* __restrict__ qn, float* __restrict__ kn) {
  int idx = blockIdx.x * blockDim.x + threadIdx.x;  // 131072 threads
  int which = idx >> 16;
  int rem = idx & 65535;
  int m = rem >> 4, h = rem & 15;
  const _Float16* p = (which ? k : q) + (size_t)m * DMOD + h * DH;
  float s = 0.0f;
#pragma unroll
  for (int d = 0; d < DH; d += 8) {
    v8h v = *(const v8h*)(p + d);
#pragma unroll
    for (int j = 0; j < 8; ++j) {
      float f = (float)v[j];
      s = fmaf(f, f, s);
    }
  }
  (which ? kn : qn)[m * NH + h] = s;
}

// --------------------------- 4) fused attention ----------------------------
// One wave = one 16-query tile of one (b,h). Streams keys in chunks of 32.
// scores: unnormalized p = exp(-arccosh(z)) = 1/(z + sqrt(z^2-1)); sum p per
// row accumulated in C-layout, reduced via shfl_xor within 16-lane halves.
__global__ __launch_bounds__(256)
void hyp_attn(const _Float16* __restrict__ qh, const _Float16* __restrict__ kh,
              const _Float16* __restrict__ vT, const float* __restrict__ qn,
              const float* __restrict__ kn, _Float16* __restrict__ oh) {
  __shared__ __align__(16) _Float16 Plds[8][16][40];   // per-wave P tile, padded

  int wave = threadIdx.x >> 5, lane = threadIdx.x & 31;
  int lo = lane & 15, hh = lane >> 4;
  int b = blockIdx.z, h = blockIdx.y;
  int q0 = (blockIdx.x * 8 + wave) * 16;               // query tile start
  int mbase = b * NSEQ;                                // row offset into [m] dim

  // --- Q A-fragments (16 x 64 split as two 16x32) ---
  const _Float16* Qb = qh + (size_t)(mbase + q0 + lo) * DMOD + h * DH;
  v16h aq[2];
#pragma unroll
  for (int s = 0; s < 2; ++s) {
    v8h l = *(const v8h*)(Qb + s * 32 + hh * 16);
    v8h g = *(const v8h*)(Qb + s * 32 + hh * 16 + 8);
    aq[s] = cat16(l, g);
  }

  // per-lane query norms for the 8 rows this lane owns in C-layout
  float qnv[8], arow[8];
  int r0 = q0 + hh * 8;
#pragma unroll
  for (int r = 0; r < 8; ++r) {
    float t = qn[(mbase + r0 + r) * NH + h];
    qnv[r] = t;
    arow[r] = 1.0f - t;                                // c = 1
  }

  v8f acc[4];
#pragma unroll
  for (int t = 0; t < 4; ++t) acc[t] = zero8f();
  float lacc[8];
#pragma unroll
  for (int r = 0; r < 8; ++r) lacc[r] = 0.0f;

  for (int kk = 0; kk < NSEQ; kk += 32) {
    // --- scores S = Q . K^T : two 16x16 tiles, contraction over Dh=64 ---
    v8f s0 = zero8f(), s1 = zero8f();
    {
      const _Float16* Kb0 = kh + (size_t)(mbase + kk + lo) * DMOD + h * DH + hh * 16;
      v16h b0 = *(const v16h*)(Kb0);
      v16h b1 = *(const v16h*)(Kb0 + 32);
      s0 = WMMA_F16(aq[0], b0, s0);
      s0 = WMMA_F16(aq[1], b1, s0);
      const _Float16* Kb1 = kh + (size_t)(mbase + kk + 16 + lo) * DMOD + h * DH + hh * 16;
      v16h b2 = *(const v16h*)(Kb1);
      v16h b3 = *(const v16h*)(Kb1 + 32);
      s1 = WMMA_F16(aq[0], b2, s1);
      s1 = WMMA_F16(aq[1], b3, s1);
    }

    // --- hyperbolic weight: p = 1/(z + sqrt(z^2-1)) = exp(-arccosh(z)) ---
    // v_rcp_f32 / v_sqrt_f32 (TRANS pipe, ~1 ulp) instead of IEEE div/sqrt:
    // the N^2 epilogue must not become VALU-bound next to the WMMA stream.
#pragma unroll
    for (int tt = 0; tt < 2; ++tt) {
      int key = kk + tt * 16 + lo;                     // this lane's column
      float knv = kn[(mbase + key) * NH + h];
      float bcol = 1.0f - knv;
#pragma unroll
      for (int r = 0; r < 8; ++r) {
        float sv     = (tt == 0) ? s0[r] : s1[r];
        float diff   = qnv[r] + knv - 2.0f * sv;       // ||q-k||^2
        float den    = fmaxf(arow[r] * bcol, 1e-6f);
        float invden = __builtin_amdgcn_rcpf(den);
        float z      = fmaxf(fmaf(2.0f * diff, invden, 1.0f), 1.0f);
        float w      = z + __builtin_amdgcn_sqrtf(fmaf(z, z, -1.0f));
        float p      = __builtin_amdgcn_rcpf(w);       // exp(-arccosh(z))
        lacc[r] += p;
        Plds[wave][hh * 8 + r][tt * 16 + lo] = (_Float16)p;
      }
    }

    // --- reload P as a 16x32 A-fragment (same-wave DS ops are in-order) ---
    v8h pl = *(const v8h*)&Plds[wave][lo][hh * 16];
    v8h pg = *(const v8h*)&Plds[wave][lo][hh * 16 + 8];
    v16h pa = cat16(pl, pg);

    // --- O += P . V  (B-frags from transposed V: columns contiguous) ---
#pragma unroll
    for (int t = 0; t < 4; ++t) {
      const _Float16* Vb =
          vT + (size_t)(h * DH + t * 16 + lo) * MROW + mbase + kk + hh * 16;
      v16h bv = *(const v16h*)Vb;
      acc[t] = WMMA_F16(pa, bv, acc[t]);
    }
  }

  // --- softmax denominators: reduce lacc across the 16-lane half ---
#pragma unroll
  for (int msk = 1; msk < 16; msk <<= 1)
#pragma unroll
    for (int r = 0; r < 8; ++r) lacc[r] += __shfl_xor(lacc[r], msk, 32);
  float invl[8];
#pragma unroll
  for (int r = 0; r < 8; ++r) invl[r] = __builtin_amdgcn_rcpf(lacc[r]);

  // --- store normalized O (f16 row-major for the final GEMM) ---
#pragma unroll
  for (int t = 0; t < 4; ++t)
#pragma unroll
    for (int r = 0; r < 8; ++r)
      oh[(size_t)(mbase + q0 + hh * 8 + r) * DMOD + h * DH + t * 16 + lo] =
          (_Float16)(acc[t][r] * invl[r]);
}

// ------------------------------- launcher ----------------------------------
extern "C" void kernel_launch(void* const* d_in, const int* in_sizes, int n_in,
                              void* d_out, int out_size, void* d_ws, size_t ws_size,
                              hipStream_t stream) {
  const float* x  = (const float*)d_in[0];
  const float* wq = (const float*)d_in[1];
  const float* wk = (const float*)d_in[2];
  const float* wv = (const float*)d_in[3];
  const float* wo = (const float*)d_in[4];
  _Float16* ws = (_Float16*)d_ws;
  float* qn = (float*)((char*)d_ws + QN_BYTE);
  float* kn = (float*)((char*)d_ws + KN_BYTE);

  // 1) f32 -> f16 (8M elements, 4 per thread)
  cvt_f32_to_f16<<<8192, 256, 0, stream>>>(x, wq, wk, wv, wo, ws);

  // 2) projections (4096 wave tiles / 8 waves per block = 512 blocks)
  gemm16<0><<<512, 256, 0, stream>>>(ws + HX, ws + HWQ, (void*)(ws + HQ));
  gemm16<0><<<512, 256, 0, stream>>>(ws + HX, ws + HWK, (void*)(ws + HK));
  gemm16<1><<<512, 256, 0, stream>>>(ws + HX, ws + HWV, (void*)(ws + HVT));

  // 3) per-head squared norms (2 * 4096 * 16 threads)
  head_norms<<<512, 256, 0, stream>>>(ws + HQ, ws + HK, qn, kn);

  // 4) fused hyperbolic attention: grid (qtiles/8, H, B)
  dim3 ag(NSEQ / (16 * 8), NH, BSZ);
  hyp_attn<<<ag, 256, 0, stream>>>(ws + HQ, ws + HK, ws + HVT, qn, kn, ws + HO);

  // 5) output projection, f32 result
  gemm16<2><<<512, 256, 0, stream>>>(ws + HO, ws + HWO, d_out);
}
